// AMM_76647986364863
// MI455X (gfx1250) — matrix operand order repaired and checked by reading, not compiled, not run on hardware
//
#include <hip/hip_runtime.h>

// ---------------------------------------------------------------------------
// MI455X (gfx1250): 16 WMMA GEMM passes (f16 in / f32 acc) with fused
// epilogues.  Tiles are DMA'd global->LDS by the Tensor Data Mover
// (tensor_load_to_lds, TENSORcnt) with hardware LDS row padding, double
// buffered.  Weights are pre-converted once per launch into f16 "B^T"
// (N x K) layout so every GEMM pass is the same TDM-fed NT kernel.
// Roofline: 32.4 GFLOP vs ~48MB unique data (weights L2-resident) ->
// matrix-core bound; V_WMMA_F32_16X16X32_F16 is the core instruction.
// ---------------------------------------------------------------------------

typedef __attribute__((ext_vector_type(16))) _Float16 v16h;
typedef __attribute__((ext_vector_type(8)))  _Float16 v8h;
typedef __attribute__((ext_vector_type(8)))  float    v8f;
typedef __attribute__((ext_vector_type(4)))  float    v4f;
typedef __attribute__((ext_vector_type(4)))  unsigned int u32x4;
typedef __attribute__((ext_vector_type(8)))  int      i32x8;
typedef __attribute__((ext_vector_type(4)))  int      i32x4;

constexpr int TILE_M = 128;
constexpr int TILE_N = 128;
constexpr int TILE_K = 32;     // one WMMA K per stage
constexpr int LDSR   = 48;     // halves per LDS row: 64B data + 32B TDM pad

// epilogue modes
constexpr int MODE_H16      = 0;  // outH = (f16)acc
constexpr int MODE_COLSCALE = 1;  // outH = (f16)(acc * scale[col])
constexpr int MODE_F32H16   = 2;  // outF = acc ; outH = (f16)acc
constexpr int MODE_RESID    = 3;  // outH = (f16)(res[o] - acc)
constexpr int MODE_SOFT     = 4;  // z = soft(zf[o] + acc, 1); zf = z ; outH = (f16)z
constexpr int MODE_ADDOUT   = 5;  // outF = zf[o] + acc

// ---------------------------------------------------------------------------
// Tensor Data Mover: DMA a (128 rows x 32 halves) tile, row stride ldk halves,
// into LDS with 32B padding after every 64B row (-> 96B LDS rows).
// D# packing per CDNA5 ISA 8.3-8.6.
// ---------------------------------------------------------------------------
__device__ __forceinline__ void tdm_load_tile_f16(const _Float16* gptr, int ldk,
                                                  unsigned lds_off) {
    unsigned long long ga = (unsigned long long)(size_t)(const void*)gptr;
    u32x4 g0 = {
        1u,                                   // count=1, user mode, no gather
        lds_off,                              // lds_addr (bytes)
        (unsigned)ga,                         // global_addr[31:0]
        ((unsigned)(ga >> 32) & 0x01FFFFFFu)  // global_addr[56:32]
            | 0x80000000u                     // type=2 ("image")
    };
    i32x8 g1 = {
        (1 << 16) | (1 << 20) | (3 << 22) | (7 << 25),
        //  data_size=2B | pad_enable | pad_interval=16DW | pad_amount=8DW
        0,                                    // atomic_addr=0, tensor_dim0[15:0]=0
        0x4000,                               // tensor_dim0[31:16]=0x4000 (huge), dim1 lo=0
        (32 << 16) | 0x4000,                  // tensor_dim1[31:16]=huge, tile_dim0=32
        128,                                  // tile_dim1=128, tile_dim2=0
        ldk,                                  // tensor_dim0_stride[31:0]
        0,                                    // stride0 hi=0, stride1 lo=0
        0
    };
    i32x4 g2 = {0, 0, 0, 0};
    i32x4 g3 = {0, 0, 0, 0};
#if __clang_major__ >= 23
    i32x8 g4 = {0, 0, 0, 0, 0, 0, 0, 0};
    __builtin_amdgcn_tensor_load_to_lds(g0, g1, g2, g3, g4, 0);
#else
    __builtin_amdgcn_tensor_load_to_lds(g0, g1, g2, g3, 0);
#endif
}

// ---------------------------------------------------------------------------
// one-time prep kernels (run each launch; deterministic)
// ---------------------------------------------------------------------------
__global__ void AMM_cvt_f32_to_f16(const float* __restrict__ in,
                                   _Float16* __restrict__ out, int n) {
    int i = (blockIdx.x * blockDim.x + threadIdx.x) * 4;
    if (i + 3 < n) {
        v4f v = *(const v4f*)(in + i);
        out[i + 0] = (_Float16)v.x;
        out[i + 1] = (_Float16)v.y;
        out[i + 2] = (_Float16)v.z;
        out[i + 3] = (_Float16)v.w;
    }
}

// in: R x C f32 row-major  ->  out: C x R f16 row-major (coalesced both sides)
__global__ __launch_bounds__(256) void AMM_transpose_cvt(
    const float* __restrict__ in, _Float16* __restrict__ out, int R, int C) {
    __shared__ float t[32][33];
    int bx = blockIdx.x * 32;              // C offset
    int by = blockIdx.y * 32;              // R offset
    int tx = threadIdx.x & 31, ty = threadIdx.x >> 5;
    #pragma unroll
    for (int i = 0; i < 32; i += 8)
        t[ty + i][tx] = in[(size_t)(by + ty + i) * C + bx + tx];
    __syncthreads();
    #pragma unroll
    for (int i = 0; i < 32; i += 8)
        out[(size_t)(bx + ty + i) * R + by + tx] = (_Float16)t[tx][ty + i];
}

// ---------------------------------------------------------------------------
// WMMA GEMM, TDM-fed, double-buffered.
// A: M x K f16 row-major.  Bp: N x K f16 row-major (pre-transposed weights).
// ---------------------------------------------------------------------------
template<int MODE>
__global__ __launch_bounds__(256) void AMM_gemm_tdm(
    const _Float16* __restrict__ A,
    const _Float16* __restrict__ Bp,
    int M, int N, int K,
    float*          __restrict__ outF,
    _Float16*       __restrict__ outH,
    const float*    __restrict__ scale,
    const float*    __restrict__ res,
    float*          __restrict__ zf)
{
    __shared__ _Float16 Alds[2][TILE_M][LDSR];
    __shared__ _Float16 Blds[2][TILE_N][LDSR];

    const int tid      = threadIdx.x;
    const int lane     = tid & 31;
    const int wave     = tid >> 5;
    const int wm       = wave & 3;            // 4 waves along M -> 32 rows each
    const int wn       = wave >> 2;           // 2 waves along N -> 64 cols each
    const int half_sel = (lane >> 4) & 1;
    const int l16      = lane & 15;

    const int m0 = blockIdx.y * TILE_M;
    const int n0 = blockIdx.x * TILE_N;
    const int steps = K >> 5;

    v8f acc[2][4] = {};

    auto issue = [&](int kt, int buf) {
        tdm_load_tile_f16(A  + (size_t)m0 * K + (kt << 5), K,
                          (unsigned)(size_t)(void*)&Alds[buf][0][0]);
        tdm_load_tile_f16(Bp + (size_t)n0 * K + (kt << 5), K,
                          (unsigned)(size_t)(void*)&Blds[buf][0][0]);
    };

    if (wave == 0) issue(0, 0);

    for (int kt = 0; kt < steps; ++kt) {
        const int buf = kt & 1;
        if (wave == 0) {
            if (kt + 1 < steps) {
                issue(kt + 1, buf ^ 1);                  // prefetch next tiles
                __builtin_amdgcn_s_wait_tensorcnt(2);    // current pair landed
            } else {
                __builtin_amdgcn_s_wait_tensorcnt(0);
            }
        }
        __syncthreads();                                 // data ready for all waves

        // ---- fragments (ISA 7.12.2 layouts) + 8 WMMAs ----
        v16h afr[2];
        #pragma unroll
        for (int mi = 0; mi < 2; ++mi) {
            // A: lanes 0-15 row m, K {0..7,16..23}; lanes 16-31: K {8..15,24..31}
            const _Float16* ap = &Alds[buf][wm * 32 + mi * 16 + l16][half_sel * 8];
            v8h lo = *(const v8h*)ap;
            v8h hi = *(const v8h*)(ap + 16);
            afr[mi] = __builtin_shufflevector(lo, hi, 0,1,2,3,4,5,6,7,8,9,10,11,12,13,14,15);
        }
        v16h bfr[4];
        #pragma unroll
        for (int ni = 0; ni < 4; ++ni) {
            // B: lanes 0-15 col n, K {0..7,16..23}; lanes 16-31: K {8..15,24..31}
            const _Float16* bp = &Blds[buf][wn * 64 + ni * 16 + l16][half_sel * 8];
            v8h lo = *(const v8h*)bp;
            v8h hi = *(const v8h*)(bp + 16);
            bfr[ni] = __builtin_shufflevector(lo, hi, 0,1,2,3,4,5,6,7,8,9,10,11,12,13,14,15);
        }
        #pragma unroll
        for (int mi = 0; mi < 2; ++mi)
            #pragma unroll
            for (int ni = 0; ni < 4; ++ni)
                acc[mi][ni] = __builtin_amdgcn_wmma_f32_16x16x32_f16(
                    false, afr[mi], false, bfr[ni],
                    (short)0, acc[mi][ni], false, false);

        __syncthreads();                                 // before buf is re-filled
    }

    // ---- fused epilogue.  C/D layout: elem r -> M = r + 8*half_sel, N = l16 ----
    #pragma unroll
    for (int mi = 0; mi < 2; ++mi) {
        #pragma unroll
        for (int ni = 0; ni < 4; ++ni) {
            int col = n0 + wn * 64 + ni * 16 + l16;
            #pragma unroll
            for (int r = 0; r < 8; ++r) {
                int row = m0 + wm * 32 + mi * 16 + r + half_sel * 8;
                size_t o = (size_t)row * N + col;
                float v = acc[mi][ni][r];
                if constexpr (MODE == MODE_H16) {
                    outH[o] = (_Float16)v;
                } else if constexpr (MODE == MODE_COLSCALE) {
                    v *= scale[col];
                    outH[o] = (_Float16)v;
                } else if constexpr (MODE == MODE_F32H16) {
                    outF[o] = v;
                    outH[o] = (_Float16)v;
                } else if constexpr (MODE == MODE_RESID) {
                    outH[o] = (_Float16)(res[o] - v);
                } else if constexpr (MODE == MODE_SOFT) {
                    float zv = zf[o] + v;
                    float a  = fabsf(zv) - 1.0f;
                    a = a > 0.0f ? a : 0.0f;
                    float s = (zv >= 0.0f) ? a : -a;
                    zf[o]  = s;
                    outH[o] = (_Float16)s;
                } else { // MODE_ADDOUT
                    outF[o] = zf[o] + v;
                }
            }
        }
    }
}

extern "C" void kernel_launch(void* const* d_in, const int* in_sizes, int n_in,
                              void* d_out, int out_size, void* d_ws, size_t ws_size,
                              hipStream_t stream) {
    (void)in_sizes; (void)n_in; (void)out_size; (void)ws_size;
    constexpr int BATCH = 512, N_ = 2048, HALF = 1024;

    const float* x         = (const float*)d_in[0];
    const float* key_enc   = (const float*)d_in[1];
    const float* val_enc   = (const float*)d_in[2];
    const float* keys_t0   = (const float*)d_in[3];
    const float* vals_t0   = (const float*)d_in[4];
    const float* scales_t0 = (const float*)d_in[5];
    const float* keys_t1   = (const float*)d_in[6];
    const float* vals_t1   = (const float*)d_in[7];
    const float* scales_t1 = (const float*)d_in[8];
    float* out = (float*)d_out;

    // ---- workspace carve-up (~38 MB) ----
    char* ws = (char*)d_ws;
    // f16 weight buffers, all N x K row-major ("B^T-ready")
    _Float16* keyencP  = (_Float16*)ws; ws += (size_t)HALF * N_   * 2; // key_enc^T
    _Float16* keyst0P  = (_Float16*)ws; ws += (size_t)HALF * HALF * 2; // keys_t0
    _Float16* valst0P  = (_Float16*)ws; ws += (size_t)HALF * HALF * 2; // vals_t0^T
    _Float16* valencTP = (_Float16*)ws; ws += (size_t)N_   * HALF * 2; // val_enc
    _Float16* valencNP = (_Float16*)ws; ws += (size_t)HALF * N_   * 2; // val_enc^T
    _Float16* keyst1P  = (_Float16*)ws; ws += (size_t)HALF * N_   * 2; // keys_t1^T
    _Float16* valst1P  = (_Float16*)ws; ws += (size_t)N_   * HALF * 2; // vals_t1
    // activations
    _Float16* x16  = (_Float16*)ws; ws += (size_t)BATCH * N_   * 2;
    _Float16* q16  = (_Float16*)ws; ws += (size_t)BATCH * HALF * 2;
    _Float16* b16  = (_Float16*)ws; ws += (size_t)BATCH * HALF * 2;
    float*    ybf  = (float*)   ws; ws += (size_t)BATCH * HALF * 4;
    _Float16* yb16 = (_Float16*)ws; ws += (size_t)BATCH * HALF * 2;
    float*    zfm  = (float*)   ws; ws += (size_t)BATCH * N_   * 4;
    _Float16* z16  = (_Float16*)ws; ws += (size_t)BATCH * N_   * 2;
    _Float16* r16  = (_Float16*)ws; ws += (size_t)BATCH * HALF * 2;

    dim3 blk(256);
    auto cgrid = [](int n) { return dim3((n / 4 + 255) / 256); };
    auto tgrid = [](int R, int C) { return dim3(C / 32, R / 32); };

    // ---- weight / input prep ----
    AMM_cvt_f32_to_f16<<<cgrid(BATCH * N_), blk, 0, stream>>>(x, x16, BATCH * N_);
    AMM_transpose_cvt<<<tgrid(N_, HALF), blk, 0, stream>>>(key_enc, keyencP, N_, HALF);
    AMM_cvt_f32_to_f16<<<cgrid(HALF * HALF), blk, 0, stream>>>(keys_t0, keyst0P, HALF * HALF);
    AMM_transpose_cvt<<<tgrid(HALF, HALF), blk, 0, stream>>>(vals_t0, valst0P, HALF, HALF);
    AMM_cvt_f32_to_f16<<<cgrid(N_ * HALF), blk, 0, stream>>>(val_enc, valencTP, N_ * HALF);
    AMM_transpose_cvt<<<tgrid(N_, HALF), blk, 0, stream>>>(val_enc, valencNP, N_, HALF);
    AMM_transpose_cvt<<<tgrid(N_, HALF), blk, 0, stream>>>(keys_t1, keyst1P, N_, HALF);
    AMM_cvt_f32_to_f16<<<cgrid(N_ * HALF), blk, 0, stream>>>(vals_t1, valst1P, N_ * HALF);

    auto grid = [](int n) { return dim3(n / TILE_N, BATCH / TILE_M); };

    // q = x @ key_enc                                   (K=2048, N=1024)
    AMM_gemm_tdm<MODE_H16><<<grid(HALF), blk, 0, stream>>>(
        x16, keyencP, BATCH, HALF, N_, nullptr, q16, nullptr, nullptr, nullptr);
    // betas = (q @ keys_t0^T) * s0[col]                 (K=1024, N=1024)
    AMM_gemm_tdm<MODE_COLSCALE><<<grid(HALF), blk, 0, stream>>>(
        q16, keyst0P, BATCH, HALF, HALF, nullptr, b16, scales_t0, nullptr, nullptr);
    // yb = betas @ vals_t0   (f32 master + f16)         (K=1024, N=1024)
    AMM_gemm_tdm<MODE_F32H16><<<grid(HALF), blk, 0, stream>>>(
        b16, valst0P, BATCH, HALF, HALF, ybf, yb16, nullptr, nullptr, nullptr);
    // z = yb @ val_enc^T     (f32 master + f16)         (K=1024, N=2048)
    AMM_gemm_tdm<MODE_F32H16><<<grid(N_), blk, 0, stream>>>(
        yb16, valencTP, BATCH, N_, HALF, zfm, z16, nullptr, nullptr, nullptr);

    // ISTA: z = soft(z + (yb - z@val_enc)@val_enc^T, 1)
    for (int it = 0; it < 5; ++it) {
        AMM_gemm_tdm<MODE_RESID><<<grid(HALF), blk, 0, stream>>>(
            z16, valencNP, BATCH, HALF, N_, nullptr, r16, nullptr, ybf, nullptr);
        AMM_gemm_tdm<MODE_SOFT><<<grid(N_), blk, 0, stream>>>(
            r16, valencTP, BATCH, N_, HALF, nullptr, z16, nullptr, nullptr, zfm);
    }

    // betas2 = (x @ keys_t1) * s1[col]                  (K=2048, N=1024)
    AMM_gemm_tdm<MODE_COLSCALE><<<grid(HALF), blk, 0, stream>>>(
        x16, keyst1P, BATCH, HALF, N_, nullptr, b16, scales_t1, nullptr, nullptr);
    // out = z + betas2 @ vals_t1^T                      (K=1024, N=2048)
    AMM_gemm_tdm<MODE_ADDOUT><<<grid(N_), blk, 0, stream>>>(
        b16, valst1P, BATCH, N_, HALF, out, nullptr, nullptr, nullptr, zfm);
}
